// MLP_NW_dist_2_att_14370960572466
// MI455X (gfx1250) — matrix operand order repaired and checked by reading, not compiled, not run on hardware
//
#include <hip/hip_runtime.h>

// ---- problem constants (match reference) ----
#define BB 4
#define NN 1024
#define MM 1024
#define HH 64

typedef __attribute__((ext_vector_type(16))) _Float16       v16h;
typedef __attribute__((ext_vector_type(8)))  _Float16       v8h;
typedef __attribute__((ext_vector_type(4)))  _Float16       v4h;
typedef __attribute__((ext_vector_type(2)))  _Float16       v2h;
typedef __attribute__((ext_vector_type(8)))  float          v8f;
typedef __attribute__((ext_vector_type(4)))  float          v4f;
typedef __attribute__((ext_vector_type(16))) unsigned short v16u;

__device__ __forceinline__ v16h habs16(v16h x)
{
    const v16u mask = {0x7FFFu,0x7FFFu,0x7FFFu,0x7FFFu,0x7FFFu,0x7FFFu,0x7FFFu,0x7FFFu,
                       0x7FFFu,0x7FFFu,0x7FFFu,0x7FFFu,0x7FFFu,0x7FFFu,0x7FFFu,0x7FFFu};
    return __builtin_bit_cast(v16h, __builtin_bit_cast(v16u, x) & mask);
}

#define CAT16(a, b) __builtin_shufflevector(a, b, 0,1,2,3,4,5,6,7,8,9,10,11,12,13,14,15)

// -------------------------------------------------------------------------
// Stage 1: tiny MLP  (x @ W1 + b1) -> relu -> (@ W2 + b2), output as f16.
// -------------------------------------------------------------------------
template <int IN>
__global__ void mlp_kernel(const float* __restrict__ x,
                           const float* __restrict__ W1, const float* __restrict__ b1,
                           const float* __restrict__ W2, const float* __restrict__ b2,
                           _Float16* __restrict__ out, int rows)
{
    int r = blockIdx.x * blockDim.x + threadIdx.x;
    if (r >= rows) return;

    float xv[IN];
#pragma unroll
    for (int i = 0; i < IN; ++i) xv[i] = x[(size_t)r * IN + i];

    float hid[HH];
#pragma unroll
    for (int h = 0; h < HH; ++h) {
        float a = b1[h];
#pragma unroll
        for (int i = 0; i < IN; ++i) a += xv[i] * W1[i * HH + h];
        hid[h] = fmaxf(a, 0.0f);
    }

    for (int j = 0; j < HH; ++j) {
        float a = b2[j];
#pragma unroll
        for (int i = 0; i < HH; ++i) a += hid[i] * W2[i * HH + j];
        out[(size_t)r * HH + j] = (_Float16)a;
    }
}

// -------------------------------------------------------------------------
// Stage 2: L1 cdist with b128 loads + packed f16 VALU.
// -------------------------------------------------------------------------
__global__ void score_kernel(const _Float16* __restrict__ Kh,
                             const _Float16* __restrict__ Qh,
                             float* __restrict__ score)
{
    int b  = blockIdx.x >> 3;
    int m  = ((blockIdx.x & 7) << 7) + threadIdx.x;
    int n0 = blockIdx.y << 8;

    const v16h* qp = (const v16h*)(Qh + ((size_t)(b * MM + m)) * HH);
    v16h q0 = qp[0], q1 = qp[1], q2 = qp[2], q3 = qp[3];

    for (int n = n0; n < n0 + 256; ++n) {
        const v16h* kp = (const v16h*)(Kh + ((size_t)(b * NN + n)) * HH);
        v16h acc = habs16(kp[0] - q0);
        acc = acc + habs16(kp[1] - q1);
        acc = acc + habs16(kp[2] - q2);
        acc = acc + habs16(kp[3] - q3);
        v8h r8 = __builtin_shufflevector(acc, acc, 0,1,2,3,4,5,6,7)
               + __builtin_shufflevector(acc, acc, 8,9,10,11,12,13,14,15);
        v4h r4 = __builtin_shufflevector(r8, r8, 0,1,2,3)
               + __builtin_shufflevector(r8, r8, 4,5,6,7);
        v2h r2 = __builtin_shufflevector(r4, r4, 0,1)
               + __builtin_shufflevector(r4, r4, 2,3);
        float s = (float)r2[0] + (float)r2[1];
        score[((size_t)(b * NN) + n) * MM + m] = s;
    }
}

// -------------------------------------------------------------------------
// Stage 3a: repack V into WMMA B-operand lane layout (one-time, 512 KB).
// -------------------------------------------------------------------------
__global__ void vpack_kernel(const _Float16* __restrict__ Vh,
                             _Float16* __restrict__ Vpack)
{
    int t = blockIdx.x * blockDim.x + threadIdx.x;   // b*1024 + c*32 + lane
    int lane = t & 31;
    int c    = (t >> 5) & 31;
    int b    = t >> 10;
    int hi   = lane >> 4;
    int lm   = lane & 15;
    int nbase = (c << 5) + (hi << 3);

    v16h o0, o1, o2, o3;
#pragma unroll
    for (int j = 0; j < 16; ++j) {
        const _Float16* vr = Vh + ((size_t)(b * NN + nbase + j + ((j >> 3) << 3))) * HH + lm;
        o0[j] = vr[0];
        o1[j] = vr[16];
        o2[j] = vr[32];
        o3[j] = vr[48];
    }
    v16h* dst = (v16h*)(Vpack + (size_t)t * 64);
    dst[0] = o0; dst[1] = o1; dst[2] = o2; dst[3] = o3;
}

// -------------------------------------------------------------------------
// Stage 3b: per-column softmax reduce + packed f16 weight store (transposed).
// -------------------------------------------------------------------------
__global__ void colreduce_pack_kernel(const float* __restrict__ score,
                                      float* __restrict__ denom,
                                      _Float16* __restrict__ wT)
{
    int x  = blockIdx.x * blockDim.x + threadIdx.x;  // b*256 + m/4
    int b  = x >> 8;
    int m4 = (x & 255) << 2;
    const v4f* sc = (const v4f*)(score + ((size_t)b << 20) + m4);

    float smin[4] = {3.4e38f, 3.4e38f, 3.4e38f, 3.4e38f};
    for (int n = 0; n < NN; ++n) {
        v4f v = sc[(size_t)n << 8];
#pragma unroll
        for (int i = 0; i < 4; ++i) smin[i] = fminf(smin[i], v[i]);
    }
    float cmax[4];
#pragma unroll
    for (int i = 0; i < 4; ++i) cmax[i] = -0.5f * smin[i] * smin[i];

    float dsum[4] = {0.f, 0.f, 0.f, 0.f};
    for (int n0 = 0; n0 < NN; n0 += 8) {
        v8h buf0, buf1, buf2, buf3;
#pragma unroll
        for (int k = 0; k < 8; ++k) {
            v4f v = sc[(size_t)(n0 + k) << 8];
            float w0 = __expf(-0.5f * v[0] * v[0] - cmax[0]); dsum[0] += w0; buf0[k] = (_Float16)w0;
            float w1 = __expf(-0.5f * v[1] * v[1] - cmax[1]); dsum[1] += w1; buf1[k] = (_Float16)w1;
            float w2 = __expf(-0.5f * v[2] * v[2] - cmax[2]); dsum[2] += w2; buf2[k] = (_Float16)w2;
            float w3 = __expf(-0.5f * v[3] * v[3] - cmax[3]); dsum[3] += w3; buf3[k] = (_Float16)w3;
        }
        *(v8h*)(wT + ((size_t)(b * MM) + m4 + 0) * NN + n0) = buf0;
        *(v8h*)(wT + ((size_t)(b * MM) + m4 + 1) * NN + n0) = buf1;
        *(v8h*)(wT + ((size_t)(b * MM) + m4 + 2) * NN + n0) = buf2;
        *(v8h*)(wT + ((size_t)(b * MM) + m4 + 3) * NN + n0) = buf3;
    }
#pragma unroll
    for (int i = 0; i < 4; ++i) denom[(b << 10) + m4 + i] = dsum[i];
}

// -------------------------------------------------------------------------
// Stage 4: numer = att @ V via v_wmma_f32_16x16x32_f16, B operands from LDS.
// Block = 8 waves sharing one (b, n-quarter): the quarter's Vpack region
// (32 KB) is staged into LDS once (coalesced global b128 -> swizzled
// ds_store_b128, lane-stride 16B => conflict-free), then each wave's B tiles
// come from ds_load_b128. A loads stay as pipelined global b128 (unique per
// wave). 4 n-quarters write 4 disjoint numer buffers (no atomics).
// -------------------------------------------------------------------------
__global__ void attv_kernel(const _Float16* __restrict__ wT,
                            const _Float16* __restrict__ Vpack,
                            float* __restrict__ numerQ)
{
    __shared__ _Float16 smem[16384];   // 32 KB: [chunk(8)][slot(8)][lane(32)*8h]

    int tid  = threadIdx.x;
    int lane = tid & 31;
    int warp = tid >> 5;
    int b    = blockIdx.x >> 5;
    int mblk = (blockIdx.x >> 2) & 7;
    int q    = blockIdx.x & 3;
    int m0   = (mblk << 7) + (warp << 4);
    int hi   = lane >> 4;
    int lm   = lane & 15;
    int m    = m0 + lm;

    // ---- stage quarter's B operands into LDS (swizzled layout) ----
    {
        const v16h* src = (const v16h*)(Vpack + ((size_t)b << 16) + ((size_t)q << 14));
#pragma unroll
        for (int i = 0; i < 4; ++i) {
            int s = (tid << 2) + i;            // linear v16h index: ((c*32+ln)*4 + t)
            v16h val = src[s];
            int c  = s >> 7;
            int ln = (s >> 2) & 31;
            int t  = s & 3;
            v8h lo  = __builtin_shufflevector(val, val, 0,1,2,3,4,5,6,7);
            v8h hi8 = __builtin_shufflevector(val, val, 8,9,10,11,12,13,14,15);
            *(v8h*)(smem + (((c << 3) + (t << 1) + 0) << 8) + (ln << 3)) = lo;
            *(v8h*)(smem + (((c << 3) + (t << 1) + 1) << 8) + (ln << 3)) = hi8;
        }
    }
    __syncthreads();

    auto btile = [&](int c, int t) -> v16h {
        const v8h* p0 = (const v8h*)(smem + (((c << 3) + (t << 1) + 0) << 8) + (lane << 3));
        const v8h* p1 = (const v8h*)(smem + (((c << 3) + (t << 1) + 1) << 8) + (lane << 3));
        return CAT16(*p0, *p1);
    };

    const _Float16* wrow = wT + ((size_t)(b * MM + m)) * NN + (q << 8) + (hi << 3);
    float* numer = numerQ + ((size_t)q * ((size_t)BB * MM * HH));

    v8f acc0 = {}, acc1 = {}, acc2 = {}, acc3 = {};

    // pipeline the (global) A loads; B comes from LDS at short latency
    v8h a0_0, a1_0, a0_1, a1_1;
    a0_0 = *(const v8h*)(wrow);
    a1_0 = *(const v8h*)(wrow + 16);

    for (int c = 0; c < 8; c += 2) {
        const _Float16* wr1 = wrow + ((c + 1) << 5);
        a0_1 = *(const v8h*)(wr1);
        a1_1 = *(const v8h*)(wr1 + 16);
        {
            v16h A = CAT16(a0_0, a1_0);
            acc0 = __builtin_amdgcn_wmma_f32_16x16x32_f16(false, A, false, btile(c, 0), (short)0, acc0, false, false);
            acc1 = __builtin_amdgcn_wmma_f32_16x16x32_f16(false, A, false, btile(c, 1), (short)0, acc1, false, false);
            acc2 = __builtin_amdgcn_wmma_f32_16x16x32_f16(false, A, false, btile(c, 2), (short)0, acc2, false, false);
            acc3 = __builtin_amdgcn_wmma_f32_16x16x32_f16(false, A, false, btile(c, 3), (short)0, acc3, false, false);
        }
        int c2 = (c + 2 < 8) ? (c + 2) : 7;
        const _Float16* wr2 = wrow + (c2 << 5);
        a0_0 = *(const v8h*)(wr2);
        a1_0 = *(const v8h*)(wr2 + 16);
        {
            v16h A = CAT16(a0_1, a1_1);
            acc0 = __builtin_amdgcn_wmma_f32_16x16x32_f16(false, A, false, btile(c + 1, 0), (short)0, acc0, false, false);
            acc1 = __builtin_amdgcn_wmma_f32_16x16x32_f16(false, A, false, btile(c + 1, 1), (short)0, acc1, false, false);
            acc2 = __builtin_amdgcn_wmma_f32_16x16x32_f16(false, A, false, btile(c + 1, 2), (short)0, acc2, false, false);
            acc3 = __builtin_amdgcn_wmma_f32_16x16x32_f16(false, A, false, btile(c + 1, 3), (short)0, acc3, false, false);
        }
    }

    int mrow = m0 + (hi << 3);
#pragma unroll
    for (int r = 0; r < 8; ++r) {
        size_t o = (((size_t)(b << 10) + mrow + r) << 6) + lm;
        numer[o]      = acc0[r];
        numer[o + 16] = acc1[r];
        numer[o + 32] = acc2[r];
        numer[o + 48] = acc3[r];
    }
}

// -------------------------------------------------------------------------
// Stage 5: out[b,m] = ((sum of 4 quarter numer)[b,m,:] . Wo) / denom + bo
// -------------------------------------------------------------------------
__global__ void proj_kernel(const float* __restrict__ numerQ,
                            const float* __restrict__ denom,
                            const float* __restrict__ Wo,
                            const float* __restrict__ bo,
                            float* __restrict__ out)
{
    const size_t QS = (size_t)BB * MM * HH;
    int x = blockIdx.x * blockDim.x + threadIdx.x;
    if (x >= BB * MM) return;
    float inv = 1.0f / denom[x];
    float a = 0.0f;
#pragma unroll
    for (int h = 0; h < HH; ++h) {
        size_t o = (size_t)x * HH + h;
        float s = numerQ[o] + numerQ[QS + o] + numerQ[2 * QS + o] + numerQ[3 * QS + o];
        a += s * Wo[h];
    }
    out[x] = a * inv + bo[0];
}

// -------------------------------------------------------------------------
extern "C" void kernel_launch(void* const* d_in, const int* in_sizes, int n_in,
                              void* d_out, int out_size, void* d_ws, size_t ws_size,
                              hipStream_t stream)
{
    const float* coords_f = (const float*)d_in[0];
    const float* values_f = (const float*)d_in[1];
    const float* coords_t = (const float*)d_in[2];
    const float* Wk1 = (const float*)d_in[3];
    const float* bk1 = (const float*)d_in[4];
    const float* Wk2 = (const float*)d_in[5];
    const float* bk2 = (const float*)d_in[6];
    const float* Wq1 = (const float*)d_in[7];
    const float* bq1 = (const float*)d_in[8];
    const float* Wq2 = (const float*)d_in[9];
    const float* bq2 = (const float*)d_in[10];
    const float* Wv1 = (const float*)d_in[11];
    const float* bv1 = (const float*)d_in[12];
    const float* Wv2 = (const float*)d_in[13];
    const float* bv2 = (const float*)d_in[14];
    const float* Wo  = (const float*)d_in[15];
    const float* bo  = (const float*)d_in[16];

    // ---- workspace layout ----
    char* ws = (char*)d_ws;
    _Float16* Kh     = (_Float16*)(ws);                                    // 512 KB
    _Float16* Qh     = (_Float16*)(ws + (512u << 10));                     // 512 KB
    _Float16* Vh     = (_Float16*)(ws + (1024u << 10));                    // 512 KB
    _Float16* Vpack  = (_Float16*)(ws + (1536u << 10));                    // 512 KB
    float*    score  = (float*)   (ws + (2048u << 10));                    // 16 MB
    _Float16* wT     = (_Float16*)(ws + (2048u << 10) + (16u << 20));      // 8 MB
    float*    den    = (float*)   (ws + (2048u << 10) + (24u << 20));      // 16 KB
    float*    numerQ = (float*)   (ws + (2048u << 10) + (24u << 20) + (16u << 10)); // 4 MB (4 buffers)

    const int rows = BB * NN;

    // Stage 1: K, Q, V MLPs (f16 outputs)
    mlp_kernel<3><<<rows / 128, 128, 0, stream>>>(coords_f, Wk1, bk1, Wk2, bk2, Kh, rows);
    mlp_kernel<3><<<rows / 128, 128, 0, stream>>>(coords_t, Wq1, bq1, Wq2, bq2, Qh, rows);
    mlp_kernel<5><<<rows / 128, 128, 0, stream>>>(values_f, Wv1, bv1, Wv2, bv2, Vh, rows);

    // Stage 2: L1 cdist -> score
    score_kernel<<<dim3(BB * 8, 4), 128, 0, stream>>>(Kh, Qh, score);

    // Stage 3a: repack V into WMMA B-operand lane layout
    vpack_kernel<<<(BB * 1024) / 128, 128, 0, stream>>>(Vh, Vpack);

    // Stage 3b: column softmax reduce + packed f16 weight matrix (transposed)
    colreduce_pack_kernel<<<BB, 256, 0, stream>>>(score, den, wT);

    // Stage 4: att @ V via LDS-fed v_wmma_f32_16x16x32_f16 (4 n-quarters)
    attv_kernel<<<BB * 32, 256, 0, stream>>>(wT, Vpack, numerQ);

    // Stage 5: combine quarters, normalize, output projection
    proj_kernel<<<(BB * MM) / 256, 256, 0, stream>>>(numerQ, den, Wo, bo, (float*)d_out);
}
